// MemorizingGPT_16587163697317
// MI455X (gfx1250) — matrix operand order repaired and checked by reading, not compiled, not run on hardware
//
#include <hip/hip_runtime.h>

// ---------------- problem constants ----------------
#define T_   2048
#define E_   1024
#define F_   3072          // 3*E
#define M_   32768
#define H_   16
#define D_   64
#define DCH  16384         // M-columns handled per block in distance kernel
#define NCH  (M_ / DCH)    // chunks over M (=2)

typedef __bf16 bf16x16 __attribute__((ext_vector_type(16)));
typedef float  v8f     __attribute__((ext_vector_type(8)));

static __device__ __forceinline__ v8f v8f_zero() {
  v8f z = {0.f,0.f,0.f,0.f,0.f,0.f,0.f,0.f};
  return z;
}

// fp32 -> bf16 round-to-nearest-even
static __device__ __forceinline__ unsigned short f2bf(float f) {
  unsigned int u = __float_as_uint(f);
  u = (u + 0x7FFFu + ((u >> 16) & 1u)) >> 16;
  return (unsigned short)u;
}

// ---------------- WMMA fragment loaders (wave32, 16x16x32 bf16) ----------------
// A-matrix 16x32 (MxK): lane l -> row m=l&15, half=l>>4.
// VGPR j holds packed K pair: k = half*16 + j*2
static __device__ __forceinline__ bf16x16 load_a_frag(const unsigned short* A, int lda) {
  int lane = threadIdx.x & 31;
  int m = lane & 15, half = lane >> 4;
  union { bf16x16 v; unsigned int u[8]; } r;
  const unsigned short* row = A + (size_t)m * lda;
#pragma unroll
  for (int j = 0; j < 8; ++j) {
    int k = (half << 4) + (j << 1);
    r.u[j] = *(const unsigned int*)(row + k);
  }
  return r.v;
}

// B-matrix 32x16 (KxN), element B[k][n] stored as W[n*ldw + k] (weight^T pattern).
// Lanes 0-15: K=0..15, lanes 16-31: K=16..31; VGPR j packs k = half*16 + 2j, 2j+1.
static __device__ __forceinline__ bf16x16 load_b_frag_nmajor(const unsigned short* W, int ldw) {
  int lane = threadIdx.x & 31;
  int n = lane & 15, half = lane >> 4;
  union { bf16x16 v; unsigned int u[8]; } r;
  const unsigned short* row = W + (size_t)n * ldw + (half << 4);
#pragma unroll
  for (int j = 0; j < 8; ++j) r.u[j] = *(const unsigned int*)(row + (j << 1));
  return r.v;
}

// B-matrix 32x16 (KxN), element B[k][n] stored as V[k*ldv + n] (memory row indexed by k).
static __device__ __forceinline__ bf16x16 load_b_frag_kmajor(const unsigned short* V, int ldv) {
  int lane = threadIdx.x & 31;
  int n = lane & 15, half = lane >> 4;
  union { bf16x16 v; unsigned short s[16]; } r;
#pragma unroll
  for (int j = 0; j < 8; ++j) {
    int k = (half << 4) + (j << 1);
    r.s[2 * j]     = V[(size_t)k * ldv + n];
    r.s[2 * j + 1] = V[(size_t)(k + 1) * ldv + n];
  }
  return r.v;
}

static __device__ __forceinline__ v8f wmma_bf16(bf16x16 a, bf16x16 b, v8f c) {
  return __builtin_amdgcn_wmma_f32_16x16x32_bf16(false, a, false, b, (short)0, c, false, false);
}

// CDNA5 async copy global->LDS (ASYNCcnt domain). ldsoff = LDS byte offset
// (low 32 bits of a generic shared pointer), gaddr = 64-bit global address.
static __device__ __forceinline__ void async_g2lds_b128(unsigned ldsoff, unsigned long long gaddr) {
  asm volatile("global_load_async_to_lds_b128 %0, %1, off" :: "v"(ldsoff), "v"(gaddr) : "memory");
}
static __device__ __forceinline__ void wait_asynccnt0() {
  asm volatile("s_wait_asynccnt 0x0" ::: "memory");
}

// ---------------- conversion / prep kernels ----------------
__global__ void k_f32_to_bf16(const float* __restrict__ src, unsigned short* __restrict__ dst, int n) {
  for (int i = blockIdx.x * blockDim.x + threadIdx.x; i < n; i += gridDim.x * blockDim.x)
    dst[i] = f2bf(src[i]);
}

// mem_db is (M, 2, E); keys are the stride-2E rows -> contiguous bf16 (M, E)
__global__ void k_memkeys_bf16(const float* __restrict__ mem_db, unsigned short* __restrict__ mk) {
  size_t total = (size_t)M_ * E_;
  for (size_t i = (size_t)blockIdx.x * blockDim.x + threadIdx.x; i < total;
       i += (size_t)gridDim.x * blockDim.x) {
    size_t m = i >> 10, e = i & 1023;
    mk[i] = f2bf(mem_db[(m << 11) + e]);
  }
}

// one wave per memory row: ||k||^2 (fp32 source for accuracy)
__global__ void k_memkey_norms(const float* __restrict__ mem_db, float* __restrict__ norms) {
  int wave = threadIdx.x >> 5, lane = threadIdx.x & 31;
  int m = blockIdx.x * 8 + wave;
  const float* row = mem_db + (size_t)m * (2 * E_);
  float s = 0.f;
  for (int e = lane; e < E_; e += 32) { float v = row[e]; s += v * v; }
#pragma unroll
  for (int off = 16; off; off >>= 1) s += __shfl_xor(s, off, 32);
  if (lane == 0) norms[m] = s;
}

// ---------------- generic NT GEMM: C[m][n] = sum_k A[m][k]*B[n][k] ----------------
// Register-pipelined: A/B fragments for k+32 are in flight while k is multiplied.
__global__ void k_gemm_nt(const unsigned short* __restrict__ A, const unsigned short* __restrict__ B,
                          float* __restrict__ C, unsigned short* __restrict__ Cb,
                          int Mrows, int Ncols, int K) {
  int wave = threadIdx.x >> 5, lane = threadIdx.x & 31;
  int row0 = (blockIdx.y * 8 + wave) * 16;
  int col0 = blockIdx.x * 64;
  if (row0 >= Mrows) return;                   // wave-uniform
  v8f acc[4];
#pragma unroll
  for (int t = 0; t < 4; ++t) acc[t] = v8f_zero();

  bf16x16 acur = load_a_frag(A + (size_t)row0 * K, K);
  bf16x16 bcur[4];
#pragma unroll
  for (int t = 0; t < 4; ++t) bcur[t] = load_b_frag_nmajor(B + (size_t)(col0 + 16 * t) * K, K);

  for (int k0 = 0; k0 < K; k0 += 32) {
    int kn = (k0 + 32 < K) ? k0 + 32 : 0;      // last iter: dummy reload, discarded
    bf16x16 anxt = load_a_frag(A + (size_t)row0 * K + kn, K);
    bf16x16 bnxt[4];
#pragma unroll
    for (int t = 0; t < 4; ++t) bnxt[t] = load_b_frag_nmajor(B + (size_t)(col0 + 16 * t) * K + kn, K);
#pragma unroll
    for (int t = 0; t < 4; ++t) acc[t] = wmma_bf16(acur, bcur[t], acc[t]);
    acur = anxt;
#pragma unroll
    for (int t = 0; t < 4; ++t) bcur[t] = bnxt[t];
  }
  int n = lane & 15, half = lane >> 4;
#pragma unroll
  for (int t = 0; t < 4; ++t)
#pragma unroll
    for (int r = 0; r < 8; ++r) {
      size_t o = (size_t)(row0 + r + 8 * half) * Ncols + col0 + 16 * t + n;
      float v = acc[t][r];
      C[o] = v;
      if (Cb) Cb[o] = f2bf(v);
    }
}

// ---------------- flash attention (causal, per-head) ----------------
__global__ void k_flash_attn(const unsigned short* __restrict__ qkvb, float* __restrict__ Y) {
  __shared__ unsigned short pstage[8][16 * 32];   // per-wave P transpose staging
  const int LDR = F_;
  int wave = threadIdx.x >> 5, lane = threadIdx.x & 31;
  int gw = blockIdx.x * 8 + wave;
  int h  = gw >> 7;           // 128 q-tiles per head
  int q0 = (gw & 127) * 16;
  const unsigned short* Qh = qkvb + h * D_;
  const unsigned short* Kh = qkvb + E_ + h * D_;
  const unsigned short* Vh = qkvb + 2 * E_ + h * D_;
  int n = lane & 15, half = lane >> 4;

  bf16x16 aq0 = load_a_frag(Qh + (size_t)q0 * LDR + 0,  LDR);
  bf16x16 aq1 = load_a_frag(Qh + (size_t)q0 * LDR + 32, LDR);

  v8f o[4];
  float mrow[8], lrow[8];
#pragma unroll
  for (int t = 0; t < 4; ++t) o[t] = v8f_zero();
#pragma unroll
  for (int r = 0; r < 8; ++r) { mrow[r] = -3.0e38f; lrow[r] = 0.f; }

  for (int kb0 = 0; kb0 <= q0 + 15; kb0 += 32) {
    v8f s[2];
#pragma unroll
    for (int nt = 0; nt < 2; ++nt) {
      int kcol = kb0 + nt * 16;
      v8f a = v8f_zero();
      a = wmma_bf16(aq0, load_b_frag_nmajor(Kh + (size_t)kcol * LDR + 0,  LDR), a);
      a = wmma_bf16(aq1, load_b_frag_nmajor(Kh + (size_t)kcol * LDR + 32, LDR), a);
      s[nt] = a;
    }
    float tmax[8];
#pragma unroll
    for (int r = 0; r < 8; ++r) {
      int row = q0 + r + 8 * half;
      float v0 = s[0][r] * 0.125f;              // 1/sqrt(64)
      float v1 = s[1][r] * 0.125f;
      if (kb0 + n      > row) v0 = -3.0e38f;
      if (kb0 + 16 + n > row) v1 = -3.0e38f;
      s[0][r] = v0; s[1][r] = v1;
      float v = fmaxf(v0, v1);
#pragma unroll
      for (int off = 1; off < 16; off <<= 1) v = fmaxf(v, __shfl_xor(v, off, 32));
      tmax[r] = v;
    }
#pragma unroll
    for (int r = 0; r < 8; ++r) {
      float mnew = fmaxf(mrow[r], tmax[r]);
      float corr = __expf(mrow[r] - mnew);
      lrow[r] *= corr;
#pragma unroll
      for (int t = 0; t < 4; ++t) o[t][r] *= corr;
      float p0 = __expf(s[0][r] - mnew);
      float p1 = __expf(s[1][r] - mnew);
      s[0][r] = p0; s[1][r] = p1;
      float ps = p0 + p1;
#pragma unroll
      for (int off = 1; off < 16; off <<= 1) ps += __shfl_xor(ps, off, 32);
      lrow[r] += ps;
      mrow[r] = mnew;
    }
    unsigned short* ps_ = pstage[wave];
#pragma unroll
    for (int r = 0; r < 8; ++r) {
      int row = r + 8 * half;
      ps_[row * 32 + n]      = f2bf(s[0][r]);
      ps_[row * 32 + 16 + n] = f2bf(s[1][r]);
    }
    bf16x16 ap = load_a_frag(ps_, 32);
#pragma unroll
    for (int t = 0; t < 4; ++t) {
      bf16x16 bv = load_b_frag_kmajor(Vh + (size_t)kb0 * LDR + t * 16, LDR);
      o[t] = wmma_bf16(ap, bv, o[t]);
    }
  }
#pragma unroll
  for (int t = 0; t < 4; ++t)
#pragma unroll
    for (int r = 0; r < 8; ++r)
      Y[(size_t)(q0 + r + 8 * half) * E_ + h * D_ + t * 16 + n] = o[t][r] / lrow[r];
}

// ---------------- distance GEMM + streaming per-row top-3 ----------------
// score(t,m) = ||k_m||^2 - 2 q_t.k_m  (||q||^2 constant per row -> irrelevant to ranking)
// Block: 16 q-rows x DCH key-columns. Q strip (16x1024 bf16, 32KB) staged in LDS via
// CDNA5 async-to-LDS; each wave computes a 16x64 tile per 512-col batch with B frags
// double-buffered in registers. All 256 threads keep private top-3 over a
// (row, 32-col-stripe) slice; one LDS merge at the end.
__global__ void k_dist_topk(const unsigned short* __restrict__ qkvb,
                            const unsigned short* __restrict__ MK,
                            const float* __restrict__ mknorm,
                            float* __restrict__ candV, int* __restrict__ candI) {
  __shared__ char smem[65536];
  unsigned short* sA   = (unsigned short*)smem;          // 16 x 1024 bf16 = 32KB
  float*          sdst = (float*)(smem + 32768);         // 16 x 512 fp32 = 32KB

  int wave = threadIdx.x >> 5, lane = threadIdx.x & 31;
  int row0 = blockIdx.y * 16;
  int chunk0 = blockIdx.x * DCH;
  int n = lane & 15, half = lane >> 4;

  // ---- async-stage the Q strip (q = first E cols of qkv) into LDS ----
  for (int i = threadIdx.x; i < 16 * 128; i += 256) {          // 2048 x b128
    int r = i >> 7, cb = i & 127;
    unsigned ldsoff = (unsigned)(unsigned long long)(sA + r * 1024 + cb * 8);
    unsigned long long g =
        (unsigned long long)(const void*)(qkvb + (size_t)(row0 + r) * F_ + cb * 8);
    async_g2lds_b128(ldsoff, g);
  }
  wait_asynccnt0();
  __syncthreads();

  // per-thread streaming top-3 over (row = tid&15, stripe = tid>>4)
  int myrow = threadIdx.x & 15, myseg = threadIdx.x >> 4;
  float bv0 = 3.4e38f, bv1 = 3.4e38f, bv2 = 3.4e38f;
  int   bi0 = 0, bi1 = 0, bi2 = 0;

  for (int c0 = chunk0; c0 < chunk0 + DCH; c0 += 512) {
    int colb = c0 + wave * 64;
    v8f acc[4];
#pragma unroll
    for (int t = 0; t < 4; ++t) acc[t] = v8f_zero();
    bf16x16 bcur[4];
#pragma unroll
    for (int t = 0; t < 4; ++t) bcur[t] = load_b_frag_nmajor(MK + (size_t)(colb + 16 * t) * E_, E_);
    for (int k0 = 0; k0 < E_; k0 += 32) {
      bf16x16 a = load_a_frag(sA + k0, 1024);                 // LDS, shared by all waves
      int kn = (k0 + 32 < E_) ? k0 + 32 : 0;
      bf16x16 bnxt[4];
#pragma unroll
      for (int t = 0; t < 4; ++t)
        bnxt[t] = load_b_frag_nmajor(MK + (size_t)(colb + 16 * t) * E_ + kn, E_);
#pragma unroll
      for (int t = 0; t < 4; ++t) acc[t] = wmma_bf16(a, bcur[t], acc[t]);
#pragma unroll
      for (int t = 0; t < 4; ++t) bcur[t] = bnxt[t];
    }
#pragma unroll
    for (int t = 0; t < 4; ++t)
#pragma unroll
      for (int r = 0; r < 8; ++r) {
        int rr = r + 8 * half;
        int cc = wave * 64 + t * 16 + n;
        sdst[rr * 512 + cc] = mknorm[colb + t * 16 + n] - 2.0f * acc[t][r];
      }
    __syncthreads();
    // scan: each thread covers 32 cols of its row
    const float* rowd = sdst + myrow * 512 + myseg * 32;
    int gibase = c0 + myseg * 32;
    for (int j = 0; j < 32; ++j) {
      float v = rowd[j];
      int gi = gibase + j;
      if (v < bv0)      { bv2=bv1; bi2=bi1; bv1=bv0; bi1=bi0; bv0=v; bi0=gi; }
      else if (v < bv1) { bv2=bv1; bi2=bi1; bv1=v;   bi1=gi; }
      else if (v < bv2) { bv2=v;   bi2=gi; }
    }
    __syncthreads();
  }

  // ---- merge 16 per-stripe partials per row (reuse sdst region) ----
  float* sCv = (float*)(smem + 32768);
  int*   sCi = (int*)(smem + 32768 + 3 * 256 * 4);
  int slot = myseg * 16 + myrow;
  sCv[slot*3+0]=bv0; sCv[slot*3+1]=bv1; sCv[slot*3+2]=bv2;
  sCi[slot*3+0]=bi0; sCi[slot*3+1]=bi1; sCi[slot*3+2]=bi2;
  __syncthreads();
  if (threadIdx.x < 16) {
    int row = threadIdx.x;
    float v0 = 3.4e38f, v1 = 3.4e38f, v2 = 3.4e38f;
    int   i0 = 0, i1 = 0, i2 = 0;
    for (int seg = 0; seg < 16; ++seg) {
      int sl = seg * 16 + row;
#pragma unroll
      for (int j = 0; j < 3; ++j) {
        float v = sCv[sl*3+j]; int gi = sCi[sl*3+j];
        if (v < v0)      { v2=v1; i2=i1; v1=v0; i1=i0; v0=v; i0=gi; }
        else if (v < v1) { v2=v1; i2=i1; v1=v;  i1=gi; }
        else if (v < v2) { v2=v;  i2=gi; }
      }
    }
    int base = ((row0 + row) * NCH + blockIdx.x) * 3;
    candV[base+0]=v0; candV[base+1]=v1; candV[base+2]=v2;
    candI[base+0]=i0; candI[base+1]=i1; candI[base+2]=i2;
  }
}

__global__ void k_topk_merge(const float* __restrict__ candV, const int* __restrict__ candI,
                             int* __restrict__ topk) {
  int row = blockIdx.x * blockDim.x + threadIdx.x;
  if (row >= T_) return;
  float bv0 = 3.4e38f, bv1 = 3.4e38f, bv2 = 3.4e38f;
  int   bi0 = 0, bi1 = 0, bi2 = 0;
  for (int c = 0; c < NCH; ++c) {
    int base = (row * NCH + c) * 3;
#pragma unroll
    for (int j = 0; j < 3; ++j) {
      float v = candV[base + j];
      int gi = candI[base + j];
      if (v < bv0)      { bv2=bv1; bi2=bi1; bv1=bv0; bi1=bi0; bv0=v; bi0=gi; }
      else if (v < bv1) { bv2=bv1; bi2=bi1; bv1=v;   bi1=gi; }
      else if (v < bv2) { bv2=v;   bi2=gi; }
    }
  }
  topk[row*3+0]=bi0; topk[row*3+1]=bi1; topk[row*3+2]=bi2;
}

// ---------------- memory attention + gate combine (tiny; VALU) ----------------
__global__ void k_mem_combine(const float* __restrict__ qkv, const float* __restrict__ mem_db,
                              const int* __restrict__ topk, const float* __restrict__ y,
                              const float* __restrict__ gate, unsigned short* __restrict__ comb) {
  int tid = blockIdx.x * blockDim.x + threadIdx.x;
  if (tid >= T_ * H_) return;
  int t = tid >> 4, h = tid & 15;
  const float* q = qkv + (size_t)t * F_ + h * D_;
  float dot[3];
  const float* mv[3];
#pragma unroll
  for (int kk = 0; kk < 3; ++kk) {
    int id = topk[t * 3 + kk];
    const float* mk = mem_db + (size_t)id * 2 * E_ + h * D_;
    mv[kk] = mk + E_;
    float s = 0.f;
    for (int d = 0; d < D_; ++d) s += q[d] * mk[d];
    dot[kk] = s * 4096.0f;     // scale_mem = E / H^(-0.5) = E*sqrt(H) = 4096
  }
  float mx = fmaxf(dot[0], fmaxf(dot[1], dot[2]));
  float e0 = __expf(dot[0]-mx), e1 = __expf(dot[1]-mx), e2 = __expf(dot[2]-mx);
  float inv = 1.0f / (e0 + e1 + e2);
  float g = gate[h];
  for (int d = 0; d < D_; ++d) {
    float mo = (e0*mv[0][d] + e1*mv[1][d] + e2*mv[2][d]) * inv;
    float yy = y[(size_t)t * E_ + h * D_ + d];
    comb[(size_t)t * E_ + h * D_ + d] = f2bf(mo * g + yy * (1.0f - g));
  }
}

// ---------------- launch ----------------
extern "C" void kernel_launch(void* const* d_in, const int* in_sizes, int n_in,
                              void* d_out, int out_size, void* d_ws, size_t ws_size,
                              hipStream_t stream) {
  (void)in_sizes; (void)n_in; (void)out_size; (void)ws_size;
  const float* x      = (const float*)d_in[0];
  const float* mem_db = (const float*)d_in[1];
  const float* W_attn = (const float*)d_in[2];
  const float* W_proj = (const float*)d_in[3];
  const float* gate   = (const float*)d_in[4];
  float* out = (float*)d_out;

  char* w = (char*)d_ws;
  size_t off = 0;
  auto alloc = [&](size_t bytes) -> char* {
    char* p = w + off;
    off = (off + bytes + 255) & ~(size_t)255;
    return p;
  };
  unsigned short* x_bf    = (unsigned short*)alloc((size_t)T_ * E_ * 2);
  unsigned short* wa_bf   = (unsigned short*)alloc((size_t)F_ * E_ * 2);
  unsigned short* wp_bf   = (unsigned short*)alloc((size_t)E_ * E_ * 2);
  unsigned short* mk_bf   = (unsigned short*)alloc((size_t)M_ * E_ * 2);
  float*          mk_nrm  = (float*)alloc((size_t)M_ * 4);
  float*          qkv_f   = (float*)alloc((size_t)T_ * F_ * 4);
  unsigned short* qkv_bf  = (unsigned short*)alloc((size_t)T_ * F_ * 2);
  float*          y_f     = (float*)alloc((size_t)T_ * E_ * 4);
  unsigned short* comb_bf = (unsigned short*)alloc((size_t)T_ * E_ * 2);
  float*          candV   = (float*)alloc((size_t)T_ * NCH * 3 * 4);
  int*            candI   = (int*)alloc((size_t)T_ * NCH * 3 * 4);
  int*            topk    = (int*)alloc((size_t)T_ * 3 * 4);

  k_f32_to_bf16<<<1024, 256, 0, stream>>>(x,      x_bf,  T_ * E_);
  k_f32_to_bf16<<<1024, 256, 0, stream>>>(W_attn, wa_bf, F_ * E_);
  k_f32_to_bf16<<<1024, 256, 0, stream>>>(W_proj, wp_bf, E_ * E_);
  k_memkeys_bf16<<<2048, 256, 0, stream>>>(mem_db, mk_bf);
  k_memkey_norms<<<M_ / 8, 256, 0, stream>>>(mem_db, mk_nrm);

  k_gemm_nt<<<dim3(F_ / 64, T_ / 128), 256, 0, stream>>>(x_bf, wa_bf, qkv_f, qkv_bf, T_, F_, E_);
  k_flash_attn<<<(T_ / 16) * H_ / 8, 256, 0, stream>>>(qkv_bf, y_f);
  k_dist_topk<<<dim3(NCH, T_ / 16), 256, 0, stream>>>(qkv_bf, mk_bf, mk_nrm, candV, candI);
  k_topk_merge<<<T_ / 256, 256, 0, stream>>>(candV, candI, topk);
  k_mem_combine<<<(T_ * H_) / 256, 256, 0, stream>>>(qkv_f, mem_db, topk, y_f, gate, comb_bf);
  k_gemm_nt<<<dim3(E_ / 64, T_ / 128), 256, 0, stream>>>(comb_bf, wp_bf, out, (unsigned short*)nullptr, T_, E_, E_);
}
